// NVFP4ScaledLinear_89378269430231
// MI455X (gfx1250) — compile-verified, matrix operand order checked by reading, not run on hardware
//
#include <hip/hip_runtime.h>
#include <stdint.h>

typedef __attribute__((ext_vector_type(16))) _Float16 v16h;
typedef __attribute__((ext_vector_type(2)))  _Float16 v2h;
typedef __attribute__((ext_vector_type(8)))  float    v8f;
typedef __attribute__((ext_vector_type(4)))  uint32_t v4u;
typedef __attribute__((ext_vector_type(4)))  int      v4i;
typedef __attribute__((ext_vector_type(8)))  int      v8i;

#define K_DIM 4096
#define N_DIM 4096
#define BM 128          // block M tile
#define BN 128          // block N tile
#define BK 32           // K step = WMMA f16 depth
#define FSTRIDE 36      // f32 row stride in TDM buffer (32 + 4 pad dwords)
#define HSTRIDE 40      // f16 row stride in staged buffers (80B -> bank spread)
#define NTHREADS 128    // 4 waves; each wave owns a 64x64 output tile

// ---- Tensor Data Mover: x[mrow:mrow+128, k:k+32] (f32) -> LDS (padded rows) ----
__device__ __forceinline__ void tdm_issue(const float* gsrc, uint32_t lds_off, int M) {
  unsigned long long ga = (unsigned long long)(uintptr_t)gsrc;
  v4u g0;
  g0[0] = 1u;                                           // count = 1 (valid D#)
  g0[1] = lds_off;                                      // lds_addr
  g0[2] = (uint32_t)ga;                                 // global_addr[31:0]
  g0[3] = (uint32_t)((ga >> 32) & 0x1FFFFFFull)         // global_addr[56:32]
        | (2u << 30);                                   // type = 2
  v8i g1;
  g1[0] = (int)((2u << 16)                              // data_size = 4B
        | (1u << 20)                                    // pad_enable
        | (4u << 22)                                    // pad_interval: every 32 dwords
        | (3u << 25));                                  // pad_amount: 4 dwords
  g1[1] = (int)((uint32_t)(K_DIM & 0xFFFF) << 16);      // tensor_dim0 lo16 @ bit48
  g1[2] = (int)((uint32_t)(K_DIM >> 16)
        | ((uint32_t)(M & 0xFFFF) << 16));              // dim0 hi | tensor_dim1 lo
  g1[3] = (int)((uint32_t)(M >> 16)
        | ((uint32_t)BK << 16));                        // dim1 hi | tile_dim0 = 32
  g1[4] = (int)BM;                                      // tile_dim1 = 128, tile_dim2 = 0
  g1[5] = (int)K_DIM;                                   // tensor_dim0_stride lo32
  g1[6] = 0;
  g1[7] = 0;
  v4i z4 = (v4i){0, 0, 0, 0};
#if defined(__clang_major__) && (__clang_major__ >= 23)
  v8i z8 = (v8i){0, 0, 0, 0, 0, 0, 0, 0};
  __builtin_amdgcn_tensor_load_to_lds(g0, g1, z4, z4, z8, 0);
#else
  __builtin_amdgcn_tensor_load_to_lds(g0, g1, z4, z4, 0);
#endif
}

// ---- v_perm LUT dequant: 8 E2M1 codes (one packed u32) -> 4 u32 (8 scaled f16) ----
// f16 encodings of |E2M1| all have zero low byte: high bytes 00,38,3C,3E,40,42,44,46.
__device__ __forceinline__ void dequant8(uint32_t nib, uint32_t sh2, uint32_t* dst) {
  const uint32_t LUTL = 0x3E3C3800u, LUTH = 0x46444240u;
  // even codes = high nibbles, odd codes = low nibbles; sign bit ORs into byte bit7
  uint32_t hbe = __builtin_amdgcn_perm(LUTH, LUTL, (nib >> 4) & 0x07070707u)
               | (nib & 0x80808080u);
  uint32_t hbo = __builtin_amdgcn_perm(LUTH, LUTL, nib & 0x07070707u)
               | ((nib << 4) & 0x80808080u);
  v2h sv = __builtin_bit_cast(v2h, sh2);
  #pragma unroll
  for (int j = 0; j < 4; ++j) {
    // dst byte0=0x00, byte1=hbe[j], byte2=0x00, byte3=hbo[j]  (sel 0x0C -> 0x00)
    uint32_t sel  = 0x040C000Cu + (uint32_t)j * 0x01000100u;
    uint32_t pair = __builtin_amdgcn_perm(hbo, hbe, sel);
    v2h pv = __builtin_bit_cast(v2h, pair);
    dst[j] = __builtin_bit_cast(uint32_t, pv * sv);     // v_pk_mul_f16
  }
}

__device__ __forceinline__ uint32_t pack4(int4 q) {     // 4 ints (one byte each) -> u32
  return (uint32_t)q.x | ((uint32_t)q.y << 8) | ((uint32_t)q.z << 16)
       | ((uint32_t)q.w << 24);
}

__device__ __forceinline__ uint32_t scale2(float s) {   // f32 scale -> packed half2
  _Float16 hs = (_Float16)s;
  uint16_t us = __builtin_bit_cast(uint16_t, hs);
  return (uint32_t)us | ((uint32_t)us << 16);
}

// 16 f16 fragment per the 16-bit WMMA operand layout: two 8-elem K-chunks.
__device__ __forceinline__ v16h load_frag(const _Float16* row, int h) {
  uint4 c0 = *(const uint4*)(row + 8 * h);
  uint4 c1 = *(const uint4*)(row + 16 + 8 * h);
  v16h v;
  uint4* p = (uint4*)&v;
  p[0] = c0;
  p[1] = c1;
  return v;
}

__global__ __launch_bounds__(NTHREADS)
void nvfp4_linear_wmma(const float* __restrict__ x,
                       const int*   __restrict__ wp,    // one byte (2 codes) per int32
                       const float* __restrict__ ws,    // per-16 block scales
                       const float* __restrict__ bias,
                       float* __restrict__ out,
                       int M) {
  __shared__ float    fbuf[2][BM * FSTRIDE];   // 2 x 18 KB raw f32 x tiles (TDM)
  __shared__ _Float16 hbuf[BM * HSTRIDE];      // 10 KB f16 A tile
  __shared__ _Float16 wbuf[BN * HSTRIDE];      // 10 KB f16 dequantized W tile

  const int lane = threadIdx.x & 31;
  const int wave = threadIdx.x >> 5;  // 4 waves
  const int wm   = wave >> 1;         // 0..1 : 64-row slab
  const int wn   = wave & 1;          // 0..1 : 64-col slab
  const int h    = lane >> 4;
  const int l16  = lane & 15;

  const int n0 = blockIdx.x * BN;
  const int m0 = blockIdx.y * BM;

  v8f acc[4][4];
  #pragma unroll
  for (int i = 0; i < 4; ++i)
    #pragma unroll
    for (int j = 0; j < 4; ++j)
      acc[i][j] = (v8f){0.f, 0.f, 0.f, 0.f, 0.f, 0.f, 0.f, 0.f};

  const int nIter = K_DIM / BK;
  if (wave == 0)
    tdm_issue(x + (size_t)m0 * K_DIM, (uint32_t)(uintptr_t)&fbuf[0][0], M);

  for (int i = 0; i < nIter; ++i) {
    const int cur = i & 1;
    const int k0  = i * BK;

    if (wave == 0) __builtin_amdgcn_s_wait_tensorcnt(0);   // tile i arrived
    __syncthreads();                                       // visible; prev reads done
    if (wave == 0 && i + 1 < nIter)                        // prefetch tile i+1 (overlaps)
      tdm_issue(x + (size_t)m0 * K_DIM + (k0 + BK),
                (uint32_t)(uintptr_t)&fbuf[cur ^ 1][0], M);

    // ---- cooperative A staging: one f32 row -> one f16 row per thread ----
    {
      const int r = threadIdx.x;                 // 0..127
      const float*    src = &fbuf[cur][r * FSTRIDE];
      _Float16*       dst = &hbuf[r * HSTRIDE];
      uint32_t ow[16];
      #pragma unroll
      for (int j = 0; j < 8; ++j) {
        float4 f = *(const float4*)(src + 4 * j);
        v2h p0 = {(_Float16)f.x, (_Float16)f.y};
        v2h p1 = {(_Float16)f.z, (_Float16)f.w};
        ow[2 * j]     = __builtin_bit_cast(uint32_t, p0);
        ow[2 * j + 1] = __builtin_bit_cast(uint32_t, p1);
      }
      #pragma unroll
      for (int j = 0; j < 4; ++j)
        *(uint4*)(dst + 8 * j) =
            (uint4){ow[4 * j], ow[4 * j + 1], ow[4 * j + 2], ow[4 * j + 3]};
    }

    // ---- cooperative W staging: one FP4 row (32 codes) -> f16 per thread ----
    {
      const int nl  = threadIdx.x;               // 0..127 local N row
      const long e0 = (long)(n0 + nl) * K_DIM + k0;
      const int* p  = wp + (e0 >> 1);
      const int4 q0 = *(const int4*)(p);
      const int4 q1 = *(const int4*)(p + 4);
      const int4 q2 = *(const int4*)(p + 8);
      const int4 q3 = *(const int4*)(p + 12);
      const uint32_t sA = scale2(ws[(e0 >> 4)]);      // block for codes 0..15
      const uint32_t sB = scale2(ws[(e0 >> 4) + 1]);  // block for codes 16..31
      uint32_t o[16];
      dequant8(pack4(q0), sA, o);
      dequant8(pack4(q1), sA, o + 4);
      dequant8(pack4(q2), sB, o + 8);
      dequant8(pack4(q3), sB, o + 12);
      _Float16* dst = &wbuf[nl * HSTRIDE];
      #pragma unroll
      for (int j = 0; j < 4; ++j)
        *(uint4*)(dst + 8 * j) =
            (uint4){o[4 * j], o[4 * j + 1], o[4 * j + 2], o[4 * j + 3]};
    }
    __syncthreads();

    // ---- fragments from LDS + 16 WMMAs per wave per k-step ----
    v16h bfrag[4];
    #pragma unroll
    for (int nt = 0; nt < 4; ++nt)
      bfrag[nt] = load_frag(&wbuf[(wn * 64 + nt * 16 + l16) * HSTRIDE], h);

    #pragma unroll
    for (int mt = 0; mt < 4; ++mt) {
      v16h a = load_frag(&hbuf[(wm * 64 + mt * 16 + l16) * HSTRIDE], h);
      #pragma unroll
      for (int nt = 0; nt < 4; ++nt) {
        acc[mt][nt] = __builtin_amdgcn_wmma_f32_16x16x32_f16(
            false, a, false, bfrag[nt], (short)0, acc[mt][nt], false, false);
      }
    }
  }

  // ---- epilogue: bias add + store per C/D layout (lane%16 = N, v + 8h = M) ----
  float bv[4];
  #pragma unroll
  for (int nt = 0; nt < 4; ++nt)
    bv[nt] = bias[n0 + wn * 64 + nt * 16 + l16];

  #pragma unroll
  for (int mt = 0; mt < 4; ++mt) {
    #pragma unroll
    for (int nt = 0; nt < 4; ++nt) {
      const int n = n0 + wn * 64 + nt * 16 + l16;
      #pragma unroll
      for (int v = 0; v < 8; ++v) {
        const int m = m0 + wm * 64 + mt * 16 + 8 * h + v;
        out[(size_t)m * N_DIM + n] = acc[mt][nt][v] + bv[nt];
      }
    }
  }
}

extern "C" void kernel_launch(void* const* d_in, const int* in_sizes, int n_in,
                              void* d_out, int out_size, void* d_ws, size_t ws_size,
                              hipStream_t stream) {
  const float* x    = (const float*)d_in[0];
  const int*   wp   = (const int*)d_in[1];
  const float* ws   = (const float*)d_in[2];
  const float* bias = (const float*)d_in[3];
  float* out = (float*)d_out;

  const int M = in_sizes[0] / K_DIM;            // 4*2048 = 8192
  dim3 grid(N_DIM / BN, M / BM);                // 32 x 64 workgroups
  nvfp4_linear_wmma<<<grid, dim3(NTHREADS), 0, stream>>>(x, wp, ws, bias, out, M);
}
